// EdgeAwareGNN_6193342841223
// MI455X (gfx1250) — matrix operand order repaired and checked by reading, not compiled, not run on hardware
//
#include <hip/hip_runtime.h>
#include <math.h>

// Problem constants (match reference)
#define B_   8
#define N_   1024
#define D_   256
#define NH_  8
#define E_   32768
#define ED_  5
#define HD_  32
#define DFF_ 512

typedef __attribute__((ext_vector_type(16))) __bf16 v16bf;
typedef __attribute__((ext_vector_type(8)))  float  v8f;

// A-operand (16x32 bf16) element -> K index mapping (CDNA5 ISA 7.12.2):
// lanes 0-15: VGPR0-3 hold K0-7, VGPR4-7 hold K16-23 ; lanes 16-31: K8-15 / K24-31.
__device__ __forceinline__ int kmap(int i, int hi) {
  int vg  = i >> 1;
  int grp = vg >> 2;
  int kk  = ((vg & 3) << 1) | (i & 1);
  return grp * 16 + hi * 8 + kk;
}

// Async global->LDS copy, 8 bytes per active lane (CDNA5 ISA).
// Generic pointer to __shared__ has the LDS offset in its low 32 bits (ISA §10.2).
__device__ __forceinline__ void async_load_b64(const void* gptr, void* lptr) {
  unsigned loff = (unsigned)(unsigned long long)lptr;
  asm volatile("global_load_async_to_lds_b64 %0, %1, off"
               :: "v"(loff), "v"(gptr) : "memory");
}
__device__ __forceinline__ void wait_async0() {
  asm volatile("s_wait_asynccnt 0x0" ::: "memory");
}

// ---------------------------------------------------------------------------
// Block GEMM core: 8 waves (blockDim 32x8). Block output macro-tile:
//   rows  m_base .. m_base+255   (8 m-iterations x 2 wave-rows x 16)
//   cols  n_blk  .. n_blk+63     (4 wave-cols x 16)
// Weight slice (64 x Kchunk=256) staged in LDS as bf16 (padded rows to spread
// DS banks). A-operand is pre-cast bf16: pure 16B loads, no in-loop converts.
// out = A . W^T, bf16 WMMA, f32 accumulate. K must be a multiple of 256.
__device__ __forceinline__ void gemm_core(const __bf16* __restrict__ A, int lda,
                                          const float* __restrict__ W, int ldw,
                                          int m_base, int n_blk, int K, v8f* acc) {
  __shared__ __bf16 ldsB[8 * 32 * 72];  // 36 KB
  const int lane = threadIdx.x;
  const int tid  = threadIdx.y * 32 + lane;
  const int lm = lane & 15, hi = lane >> 4;
  const int ni  = threadIdx.y & 3;   // wave's n-tile
  const int mi2 = threadIdx.y >> 2;  // wave's row half (+0 / +16)

  for (int kc = 0; kc < K; kc += 256) {
    // stage W[n_blk..+63][kc..+255] -> ldsB[ks][kl][n]   (coalesced global reads)
    for (int idx = tid; idx < 64 * 256; idx += 256) {
      int n = idx >> 8;
      int k = idx & 255;
      float w = W[(size_t)(n_blk + n) * ldw + kc + k];
      ldsB[((k >> 5) * 32 + (k & 31)) * 72 + n] = (__bf16)w;
    }
    __syncthreads();
#pragma unroll
    for (int mi = 0; mi < 8; ++mi) {
      const __bf16* arow = A + (size_t)(m_base + mi * 32 + mi2 * 16 + lm) * lda + kc;
#pragma unroll
      for (int ks = 0; ks < 8; ++ks) {
        v16bf a;
#pragma unroll
        for (int i = 0; i < 16; ++i) a[i] = arow[ks * 32 + kmap(i, hi)];
        v16bf bb;
#pragma unroll
        for (int i = 0; i < 16; ++i) bb[i] = ldsB[(ks * 32 + kmap(i, hi)) * 72 + ni * 16 + lm];
        acc[mi] = __builtin_amdgcn_wmma_f32_16x16x32_bf16(false, a, false, bb, (short)0,
                                                          acc[mi], false, false);
      }
    }
    __syncthreads();
  }
}

// ---------------------------------------------------------------- fp32 -> bf16 cast
__global__ __launch_bounds__(256) void cast_kernel(const float* __restrict__ src,
                                                   __bf16* __restrict__ dst, int n) {
  int t = blockIdx.x * blockDim.x + threadIdx.x;
  if (t < n) dst[t] = (__bf16)src[t];
}

// ---------------------------------------------------------------- edge bias scatter
__global__ __launch_bounds__(256) void edge_scatter_kernel(
    const float* __restrict__ edge_feat, const float* __restrict__ edge_valid,
    const int* __restrict__ src_c, const int* __restrict__ dst_c,
    const float* __restrict__ We, float* __restrict__ bias) {
  int t = blockIdx.x * blockDim.x + threadIdx.x;  // over B*E
  if (t >= B_ * E_) return;
  int b = t / E_;
  float ev = edge_valid[t];
  int s = src_c[t], d = dst_c[t];
  float ef[ED_];
#pragma unroll
  for (int c = 0; c < ED_; ++c) ef[c] = edge_feat[(size_t)t * ED_ + c];
#pragma unroll
  for (int h = 0; h < NH_; ++h) {
    float r = 0.f;
#pragma unroll
    for (int c = 0; c < ED_; ++c) r += ef[c] * We[h * ED_ + c];
    r *= ev;
    atomicAdd(&bias[(((size_t)(b * NH_ + h) * N_ + d) * N_) + s], r);
  }
}

// ---------------------------------------------------------------- fused QKV projection
__global__ __launch_bounds__(256) void qkv_kernel(
    const __bf16* __restrict__ xb,
    const float* __restrict__ Wq, const float* __restrict__ bq,
    const float* __restrict__ Wk, const float* __restrict__ bk,
    const float* __restrict__ Wv, const float* __restrict__ bv,
    __bf16* __restrict__ qbuf, __bf16* __restrict__ kTbuf, __bf16* __restrict__ vbuf) {
  const int which = blockIdx.z;
  const float* W  = (which == 0) ? Wq : (which == 1) ? Wk : Wv;
  const float* bb = (which == 0) ? bq : (which == 1) ? bk : bv;
  const int m_base = blockIdx.x * 256;
  const int n_blk  = blockIdx.y * 64;
  v8f acc[8];
#pragma unroll
  for (int mi = 0; mi < 8; ++mi) acc[mi] = (v8f){};
  gemm_core(xb, D_, W, D_, m_base, n_blk, D_, acc);

  const int lane = threadIdx.x;
  const int lm = lane & 15, hi = lane >> 4;
  const int ni = threadIdx.y & 3, mi2 = threadIdx.y >> 2;
  const int n_g = n_blk + ni * 16 + lm;
  const int h = n_g >> 5, hd = n_g & 31;
  const float bval = bb[n_g];
#pragma unroll
  for (int mi = 0; mi < 8; ++mi) {
#pragma unroll
    for (int r = 0; r < 8; ++r) {
      int m_g = m_base + mi * 32 + mi2 * 16 + r + hi * 8;
      int b = m_g >> 10, nrow = m_g & 1023;
      __bf16 bv16 = (__bf16)(acc[mi][r] + bval);
      size_t bh = (size_t)(b * NH_ + h);
      if (which == 0)      qbuf[(bh * N_ + nrow) * HD_ + hd] = bv16;
      else if (which == 1) kTbuf[(bh * HD_ + hd) * N_ + nrow] = bv16;  // transposed
      else                 vbuf[(bh * N_ + nrow) * HD_ + hd] = bv16;
    }
  }
}

// ---------------------------------------------------------------- flash attention
// 8 waves / block, all on the same (b,h); K/V 32x32 tiles staged once per block
// per j-step via async global->LDS loads (ASYNCcnt path). Output written bf16
// (it is only consumed as the A-operand of the Wo GEMM).
__global__ __launch_bounds__(256) void attn_kernel(
    const __bf16* __restrict__ q, const __bf16* __restrict__ kT,
    const __bf16* __restrict__ v, const float* __restrict__ bias,
    const unsigned char* __restrict__ nmask, __bf16* __restrict__ outp) {
  const int wy = threadIdx.y;               // wave id 0..7
  const int lane = threadIdx.x, lm = lane & 15, hi = lane >> 4;
  const int h = blockIdx.y, b = blockIdx.z;
  const int bh = b * NH_ + h;
  const int m0 = blockIdx.x * 128 + wy * 16;

  __shared__ __bf16 ldsK[32 * 40];          // padded rows (80B) for DS banking
  __shared__ __bf16 ldsV[32 * 40];
  __shared__ float sp[8][16][33];

  const int tid = wy * 32 + lane;
  const int sk = tid >> 3;                  // staging row 0..31
  const int sj = (tid & 7) * 4;             // staging col (4 bf16 = 8 bytes)

  // Q A-operand (16 x HD), loaded once
  v16bf aq;
  const __bf16* qb = q + ((size_t)bh * N_ + m0) * HD_;
#pragma unroll
  for (int i = 0; i < 16; ++i) aq[i] = qb[lm * HD_ + kmap(i, hi)];

  v8f o0 = {}, o1 = {};
  float mrun[8], lrun[8];
#pragma unroll
  for (int r = 0; r < 8; ++r) { mrun[r] = -1e30f; lrun[r] = 0.f; }

  const float scale = 0.1767766953f;  // HD^-0.5
  const float* biasrow = bias + ((size_t)bh * N_ + m0) * N_;
  const unsigned char* mk = nmask + b * N_;
  const __bf16* kbase = kT + (size_t)bh * HD_ * N_;
  const __bf16* vbase = v + (size_t)bh * N_ * HD_;
  const v8f zero = {};

  for (int j0 = 0; j0 < N_; j0 += 32) {
    __syncthreads();  // previous iteration's LDS reads complete
    async_load_b64(kbase + (size_t)sk * N_ + j0 + sj, &ldsK[sk * 40 + sj]);
    async_load_b64(vbase + (size_t)(j0 + sk) * HD_ + sj, &ldsV[sk * 40 + sj]);
    if (j0 + 64 <= N_)
      __builtin_prefetch(&biasrow[(size_t)(hi * 8) * N_ + j0 + 32 + lm], 0, 1);
    wait_async0();
    __syncthreads();  // K/V tiles staged

    v16bf b0, b1;
#pragma unroll
    for (int i = 0; i < 16; ++i) {
      b0[i] = ldsK[kmap(i, hi) * 40 + lm];
      b1[i] = ldsK[kmap(i, hi) * 40 + 16 + lm];
    }
    v8f s0 = __builtin_amdgcn_wmma_f32_16x16x32_bf16(false, aq, false, b0, (short)0, zero, false, false);
    v8f s1 = __builtin_amdgcn_wmma_f32_16x16x32_bf16(false, aq, false, b1, (short)0, zero, false, false);

    float msk0 = mk[j0 + lm] ? 0.f : -1e30f;
    float msk1 = mk[j0 + 16 + lm] ? 0.f : -1e30f;
#pragma unroll
    for (int r = 0; r < 8; ++r) {
      int m = r + hi * 8;
      float t0 = s0[r] * scale + biasrow[(size_t)m * N_ + j0 + lm] + msk0;
      float t1 = s1[r] * scale + biasrow[(size_t)m * N_ + j0 + 16 + lm] + msk1;
      float mx = fmaxf(t0, t1);
#pragma unroll
      for (int off = 1; off < 16; off <<= 1) mx = fmaxf(mx, __shfl_xor(mx, off, 32));
      float mnew = fmaxf(mrun[r], mx);
      float alpha = __expf(mrun[r] - mnew);
      float p0 = __expf(t0 - mnew);
      float p1 = __expf(t1 - mnew);
      float rs = p0 + p1;
#pragma unroll
      for (int off = 1; off < 16; off <<= 1) rs += __shfl_xor(rs, off, 32);
      lrun[r] = lrun[r] * alpha + rs;
      mrun[r] = mnew;
      o0[r] *= alpha;
      o1[r] *= alpha;
      sp[wy][m][lm] = p0;
      sp[wy][m][16 + lm] = p1;
    }
    __syncthreads();  // sp visible / ordered before transpose read

    v16bf ap;
#pragma unroll
    for (int i = 0; i < 16; ++i) ap[i] = (__bf16)sp[wy][lm][kmap(i, hi)];
    v16bf bv0, bv1;
#pragma unroll
    for (int i = 0; i < 16; ++i) {
      bv0[i] = ldsV[kmap(i, hi) * 40 + lm];
      bv1[i] = ldsV[kmap(i, hi) * 40 + 16 + lm];
    }
    o0 = __builtin_amdgcn_wmma_f32_16x16x32_bf16(false, ap, false, bv0, (short)0, o0, false, false);
    o1 = __builtin_amdgcn_wmma_f32_16x16x32_bf16(false, ap, false, bv1, (short)0, o1, false, false);
  }

#pragma unroll
  for (int r = 0; r < 8; ++r) {
    int m = m0 + r + hi * 8;
    float inv = 1.0f / lrun[r];
    size_t base = ((size_t)(b * N_ + m)) * D_ + h * HD_;
    outp[base + lm] = (__bf16)(o0[r] * inv);
    outp[base + 16 + lm] = (__bf16)(o1[r] * inv);
  }
}

// ---------------------------------------------------------------- Wo projection + residual (pre-LN)
__global__ __launch_bounds__(256) void wo_kernel(
    const __bf16* __restrict__ attnob, const float* __restrict__ Wo,
    const float* __restrict__ bo, const float* __restrict__ x,
    float* __restrict__ x1) {
  const int m_base = blockIdx.x * 256, n_blk = blockIdx.y * 64;
  v8f acc[8];
#pragma unroll
  for (int mi = 0; mi < 8; ++mi) acc[mi] = (v8f){};
  gemm_core(attnob, D_, Wo, D_, m_base, n_blk, D_, acc);
  const int lane = threadIdx.x, lm = lane & 15, hi = lane >> 4;
  const int ni = threadIdx.y & 3, mi2 = threadIdx.y >> 2;
  const int n_g = n_blk + ni * 16 + lm;
  const float bval = bo[n_g];
#pragma unroll
  for (int mi = 0; mi < 8; ++mi)
#pragma unroll
    for (int r = 0; r < 8; ++r) {
      int m_g = m_base + mi * 32 + mi2 * 16 + r + hi * 8;
      x1[(size_t)m_g * D_ + n_g] = acc[mi][r] + bval + x[(size_t)m_g * D_ + n_g];
    }
}

// ---------------------------------------------------------------- FFN up (ReLU, bf16 out)
__global__ __launch_bounds__(256) void ffn1_kernel(
    const __bf16* __restrict__ x1b, const float* __restrict__ W1,
    const float* __restrict__ b1, __bf16* __restrict__ hbf) {
  const int m_base = blockIdx.x * 256, n_blk = blockIdx.y * 64;
  v8f acc[8];
#pragma unroll
  for (int mi = 0; mi < 8; ++mi) acc[mi] = (v8f){};
  gemm_core(x1b, D_, W1, D_, m_base, n_blk, D_, acc);
  const int lane = threadIdx.x, lm = lane & 15, hi = lane >> 4;
  const int ni = threadIdx.y & 3, mi2 = threadIdx.y >> 2;
  const int n_g = n_blk + ni * 16 + lm;
  const float bval = b1[n_g];
#pragma unroll
  for (int mi = 0; mi < 8; ++mi)
#pragma unroll
    for (int r = 0; r < 8; ++r) {
      int m_g = m_base + mi * 32 + mi2 * 16 + r + hi * 8;
      hbf[(size_t)m_g * DFF_ + n_g] = (__bf16)fmaxf(acc[mi][r] + bval, 0.f);
    }
}

// ---------------------------------------------------------------- FFN down + residual (pre-LN)
__global__ __launch_bounds__(256) void ffn2_kernel(
    const __bf16* __restrict__ hbf, const float* __restrict__ W2,
    const float* __restrict__ b2, const float* __restrict__ x1,
    float* __restrict__ outp) {
  const int m_base = blockIdx.x * 256, n_blk = blockIdx.y * 64;
  v8f acc[8];
#pragma unroll
  for (int mi = 0; mi < 8; ++mi) acc[mi] = (v8f){};
  gemm_core(hbf, DFF_, W2, DFF_, m_base, n_blk, DFF_, acc);
  const int lane = threadIdx.x, lm = lane & 15, hi = lane >> 4;
  const int ni = threadIdx.y & 3, mi2 = threadIdx.y >> 2;
  const int n_g = n_blk + ni * 16 + lm;
  const float bval = b2[n_g];
#pragma unroll
  for (int mi = 0; mi < 8; ++mi)
#pragma unroll
    for (int r = 0; r < 8; ++r) {
      int m_g = m_base + mi * 32 + mi2 * 16 + r + hi * 8;
      outp[(size_t)m_g * D_ + n_g] = acc[mi][r] + bval + x1[(size_t)m_g * D_ + n_g];
    }
}

// ---------------------------------------------------------------- row LayerNorm
// Optional node-mask multiply; optional secondary bf16 output (feeds next GEMM).
__global__ __launch_bounds__(256) void ln_kernel(
    const float* __restrict__ src, float* __restrict__ dst,
    const float* __restrict__ g, const float* __restrict__ be,
    const unsigned char* __restrict__ mask, __bf16* __restrict__ dst2) {
  const int row = blockIdx.x, tid = threadIdx.x;
  __shared__ float red[256];
  float v = src[(size_t)row * D_ + tid];
  red[tid] = v;
  __syncthreads();
  for (int s = 128; s > 0; s >>= 1) {
    if (tid < s) red[tid] += red[tid + s];
    __syncthreads();
  }
  float mean = red[0] * (1.0f / D_);
  __syncthreads();
  float dv = v - mean;
  red[tid] = dv * dv;
  __syncthreads();
  for (int s = 128; s > 0; s >>= 1) {
    if (tid < s) red[tid] += red[tid + s];
    __syncthreads();
  }
  float var = red[0] * (1.0f / D_);
  float y = dv * rsqrtf(var + 1e-5f) * g[tid] + be[tid];
  if (mask) {
    int b = row >> 10, n = row & 1023;
    y *= mask[b * N_ + n] ? 1.f : 0.f;
  }
  dst[(size_t)row * D_ + tid] = y;
  if (dst2) dst2[(size_t)row * D_ + tid] = (__bf16)y;
}

// ---------------------------------------------------------------- host launch
extern "C" void kernel_launch(void* const* d_in, const int* in_sizes, int n_in,
                              void* d_out, int out_size, void* d_ws, size_t ws_size,
                              hipStream_t stream) {
  const float* x          = (const float*)d_in[0];
  const float* edge_feat  = (const float*)d_in[1];
  const float* edge_valid = (const float*)d_in[2];
  const unsigned char* node_mask = (const unsigned char*)d_in[3];  // jnp bool_: 1 byte
  const int* src_c = (const int*)d_in[4];
  const int* dst_c = (const int*)d_in[5];
  const float* Wq = (const float*)d_in[6];  const float* bq = (const float*)d_in[7];
  const float* Wk = (const float*)d_in[8];  const float* bk = (const float*)d_in[9];
  const float* Wv = (const float*)d_in[10]; const float* bv = (const float*)d_in[11];
  const float* Wo = (const float*)d_in[12]; const float* bo = (const float*)d_in[13];
  const float* We = (const float*)d_in[14];
  const float* W1 = (const float*)d_in[15]; const float* b1 = (const float*)d_in[16];
  const float* W2 = (const float*)d_in[17]; const float* b2 = (const float*)d_in[18];
  const float* g1 = (const float*)d_in[19]; const float* be1 = (const float*)d_in[20];
  const float* g2 = (const float*)d_in[21]; const float* be2 = (const float*)d_in[22];
  float* outp = (float*)d_out;

  char* ws = (char*)d_ws;
  const size_t MB = 1u << 20;
  __bf16* qbuf   = (__bf16*)(ws + 0 * MB);   // 4 MiB  [B,NH,N,HD] bf16
  __bf16* kTbuf  = (__bf16*)(ws + 4 * MB);   // 4 MiB  [B,NH,HD,N] bf16
  __bf16* vbuf   = (__bf16*)(ws + 8 * MB);   // 4 MiB  [B,NH,N,HD] bf16
  __bf16* hbf    = (__bf16*)(ws + 12 * MB);  // 8 MiB  [B*N,DFF]   bf16
  float*  x1     = (float*)(ws + 20 * MB);   // 8 MiB  [B*N,D]     f32 (post-LN1)
  __bf16* attnob = (__bf16*)(ws + 28 * MB);  // 4 MiB  [B,N,D]     bf16
  __bf16* xb     = (__bf16*)(ws + 32 * MB);  // 4 MiB  [B,N,D]     bf16
  __bf16* x1b    = (__bf16*)(ws + 36 * MB);  // 4 MiB  [B,N,D]     bf16
  float*  bias   = (float*)(ws + 40 * MB);   // 256 MiB [B,NH,N,N] f32

  const size_t bias_bytes = (size_t)B_ * NH_ * N_ * N_ * sizeof(float);
  hipMemsetAsync(bias, 0, bias_bytes, stream);

  const int ND = B_ * N_ * D_;
  cast_kernel<<<(ND + 255) / 256, 256, 0, stream>>>(x, xb, ND);

  edge_scatter_kernel<<<(B_ * E_ + 255) / 256, 256, 0, stream>>>(
      edge_feat, edge_valid, src_c, dst_c, We, bias);

  dim3 blk(32, 8);
  qkv_kernel<<<dim3(32, 4, 3), blk, 0, stream>>>(xb, Wq, bq, Wk, bk, Wv, bv,
                                                 qbuf, kTbuf, vbuf);

  attn_kernel<<<dim3(N_ / 128, NH_, B_), blk, 0, stream>>>(qbuf, kTbuf, vbuf, bias,
                                                           node_mask, attnob);

  wo_kernel<<<dim3(32, 4), blk, 0, stream>>>(attnob, Wo, bo, x, x1);

  ln_kernel<<<B_ * N_, 256, 0, stream>>>(x1, x1, g1, be1, nullptr, x1b);

  ffn1_kernel<<<dim3(32, 8), blk, 0, stream>>>(x1b, W1, b1, hbf);

  ffn2_kernel<<<dim3(32, 4), blk, 0, stream>>>(hbf, W2, b2, x1, outp);

  ln_kernel<<<B_ * N_, 256, 0, stream>>>(outp, outp, g2, be2, node_mask, nullptr);
}
